// GNN_Classifier_21766894256130
// MI455X (gfx1250) — matrix operand order repaired and checked by reading, not compiled, not run on hardware
//
#include <hip/hip_runtime.h>

#define N_NODES   50000
#define N_EDGES   800000
#define D_IN      768
#define D_H       256
#define N_CLASSES 4
#define N_LAYERS  3
#define N_GRAPHS  64

typedef __attribute__((ext_vector_type(16))) __bf16 v16bf;
typedef __attribute__((ext_vector_type(8)))  float  v8f;

union FragA { v16bf v; unsigned short s[16]; };
union FragB { v16bf v; uint4 q[2]; };

__device__ __forceinline__ unsigned short f32_to_bf16(float f) {
  unsigned int u = __float_as_uint(f);
  u += 0x7FFFu + ((u >> 16) & 1u);       // round-to-nearest-even
  return (unsigned short)(u >> 16);
}

// Async DMA: 16 bytes per lane, global -> LDS, tracked by ASYNCcnt.
__device__ __forceinline__ void async_copy_b128(unsigned lds_off, const float* gaddr) {
  asm volatile("global_load_async_to_lds_b128 %0, %1, off"
               :: "v"(lds_off), "v"(gaddr)
               : "memory");
}
__device__ __forceinline__ void wait_async0() {
  asm volatile("s_wait_asynccnt 0" ::: "memory");
}

// ---------------------------------------------------------------------------
// Weight convert+transpose: W (K x N, f32 row-major) -> Wt (N x K, bf16)
// ---------------------------------------------------------------------------
__global__ void k_transpose_bf16(const float* __restrict__ W,
                                 unsigned short* __restrict__ Wt,
                                 int K, int N) {
  int idx = blockIdx.x * blockDim.x + threadIdx.x;
  if (idx >= K * N) return;
  int k = idx / N, n = idx - k * N;
  Wt[(size_t)n * K + k] = f32_to_bf16(W[idx]);
}

// ---------------------------------------------------------------------------
// Degree / norm
// ---------------------------------------------------------------------------
__global__ void k_deg_init(float* deg) {
  int i = blockIdx.x * blockDim.x + threadIdx.x;
  if (i < N_NODES) deg[i] = 1.0f;                       // self-loop
}
__global__ void k_deg_edges(const int* __restrict__ dst, float* __restrict__ deg) {
  int e = blockIdx.x * blockDim.x + threadIdx.x;
  if (e < N_EDGES) atomicAdd(&deg[dst[e]], 1.0f);
}
__global__ void k_dinv(const float* __restrict__ deg, float* __restrict__ dinv) {
  int i = blockIdx.x * blockDim.x + threadIdx.x;
  if (i < N_NODES) dinv[i] = rsqrtf(deg[i]);
}

// ---------------------------------------------------------------------------
// WMMA GEMM:  C[M x 256] = A[M x K](f32) * Bt[256 x K](bf16, pre-transposed)
//             (+ optional bias, optional relu).  M % 16 == 0, K % 32 == 0.
// 128 threads (4 waves) own one 16-row M-tile x all 256 columns.
// A slabs (16x32 f32, 2 KB) are staged with double-buffered async DMA
// (global_load_async_to_lds_b128, ASYNCcnt); f32 -> bf16 happens in the
// per-lane fragment build from LDS.
// ---------------------------------------------------------------------------
__global__ __launch_bounds__(128)
void k_gemm_bf16_n256(const float* __restrict__ A,
                      const unsigned short* __restrict__ Bt,
                      const float* __restrict__ bias,
                      float* __restrict__ C,
                      int K, int relu) {
  __shared__ float Asf[2][16 * 32];      // two 2 KB f32 slabs

  const int  tid  = threadIdx.x;
  const int  wave = tid >> 5;
  const int  lane = tid & 31;
  const int  g    = lane >> 4;           // half-wave group (0/1)
  const int  lr   = lane & 15;
  const long m0   = (long)blockIdx.x * 16;
  const int  nbase = wave * 64;

  const int e4  = tid << 2;              // this thread's 4-float chunk of slab
  const int row = e4 >> 5;
  const int col = e4 & 31;
  const float* gA = A + (m0 + row) * K + col;

  v8f acc[4] = {};

  // Prologue: async-stage slab kk=0 into buffer 0
  async_copy_b128((unsigned)(size_t)&Asf[0][e4], gA);

  int nbuf = 1;
  for (int kk = 0; kk < K; kk += 32) {
    const int cur = nbuf ^ 1;
    wait_async0();                       // my piece of slab `cur` has landed
    __syncthreads();                     // everyone's pieces landed; readers of
                                         // buffer `nbuf` (iter kk-32) are done
    if (kk + 32 < K)                     // prefetch next slab via async DMA
      async_copy_b128((unsigned)(size_t)&Asf[nbuf][e4], gA + kk + 32);

    // A fragment (ISA 16-bit A 16x32 layout): lane holds row lr,
    // K = g*8+0..7 in VGPRs 0-3 and K = 16+g*8+0..7 in VGPRs 4-7.
    FragA a;
    {
      const float* arow = &Asf[cur][(lr << 5)];
#pragma unroll
      for (int j = 0; j < 8; ++j) {
        a.s[j]     = f32_to_bf16(arow[(g << 3) + j]);
        a.s[8 + j] = f32_to_bf16(arow[16 + (g << 3) + j]);
      }
    }

#pragma unroll
    for (int nt = 0; nt < 4; ++nt) {
      // B fragment (ISA 16-bit B 32x16 layout): lane holds column lr,
      // K = g*8+0..7 in elements 0-7 and K = 16+g*8+0..7 in elements 8-15.
      FragB b;
      const unsigned short* bp =
          Bt + (size_t)(nbase + nt * 16 + lr) * K + kk + (g << 3);
      b.q[0] = *(const uint4*)bp;
      b.q[1] = *(const uint4*)(bp + 16);
      if (kk + 32 < K) __builtin_prefetch(bp + 32, 0, 3);  // global_prefetch_b8
      acc[nt] = __builtin_amdgcn_wmma_f32_16x16x32_bf16(
          false, a.v, false, b.v, (short)0, acc[nt], false, false);
    }
    nbuf ^= 1;
  }

  // D layout: VGPR r -> row m0 + r + 8*g, col = nbase + nt*16 + lr
#pragma unroll
  for (int nt = 0; nt < 4; ++nt) {
    const int colo = nbase + nt * 16 + lr;
    const float bv = bias ? bias[colo] : 0.0f;
#pragma unroll
    for (int r = 0; r < 8; ++r) {
      float v = acc[nt][r] + bv;
      if (relu) v = fmaxf(v, 0.0f);
      C[(m0 + r + (g << 3)) * D_H + colo] = v;
    }
  }
}

// ---------------------------------------------------------------------------
// Aggregation
// ---------------------------------------------------------------------------
__global__ void k_acc_init(const float* __restrict__ convB, float* __restrict__ acc) {
  int idx = blockIdx.x * blockDim.x + threadIdx.x;
  if (idx < N_NODES * D_H) acc[idx] = convB[idx & (D_H - 1)];
}

// One wave per edge: gather m[src]*norm, scatter-add to acc[dst] (L2 atomics)
__global__ void k_aggregate(const int* __restrict__ src, const int* __restrict__ dst,
                            const float* __restrict__ dinv,
                            const float* __restrict__ m, float* __restrict__ acc) {
  const int e    = (blockIdx.x * blockDim.x + threadIdx.x) >> 5;
  const int lane = threadIdx.x & 31;
  if (e >= N_EDGES) return;
  const int s = src[e], d = dst[e];
  const float nrm = dinv[s] * dinv[d];
  const float4* mrow = (const float4*)(m + (size_t)s * D_H);
  float* arow = acc + (size_t)d * D_H;
#pragma unroll
  for (int j = 0; j < 2; ++j) {
    const int q = lane + j * 32;
    float4 v = mrow[q];
    atomicAdd(arow + q * 4 + 0, v.x * nrm);
    atomicAdd(arow + q * 4 + 1, v.y * nrm);
    atomicAdd(arow + q * 4 + 2, v.z * nrm);
    atomicAdd(arow + q * 4 + 3, v.w * nrm);
  }
}

// Self-loop + residual + ReLU + LayerNorm fused; wave-per-row (wave32)
__global__ __launch_bounds__(256)
void k_combine_ln(const float* __restrict__ accA, const float* __restrict__ m,
                  const float* __restrict__ dinv, float* __restrict__ h,
                  const float* __restrict__ lnG, const float* __restrict__ lnB) {
  const int row  = blockIdx.x * 8 + (threadIdx.x >> 5);
  const int lane = threadIdx.x & 31;
  if (row >= N_NODES) return;
  const float di = dinv[row];
  const float d2 = di * di;                       // self-loop norm
  const size_t base = (size_t)row * D_H;
  float vals[8];
  float sum = 0.f, sq = 0.f;
#pragma unroll
  for (int j = 0; j < 8; ++j) {
    const int c = lane + j * 32;
    float t = accA[base + c] + m[base + c] * d2 + h[base + c];
    t = fmaxf(t, 0.f);
    vals[j] = t;
    sum += t; sq += t * t;
  }
#pragma unroll
  for (int off = 16; off > 0; off >>= 1) {
    sum += __shfl_xor(sum, off, 32);
    sq  += __shfl_xor(sq,  off, 32);
  }
  const float mean = sum * (1.f / D_H);
  const float var  = sq  * (1.f / D_H) - mean * mean;
  const float rstd = rsqrtf(var + 1e-5f);
#pragma unroll
  for (int j = 0; j < 8; ++j) {
    const int c = lane + j * 32;
    h[base + c] = (vals[j] - mean) * rstd * lnG[c] + lnB[c];
  }
}

// ---------------------------------------------------------------------------
// Global max pool (segment_max over sorted batch ids)
// ---------------------------------------------------------------------------
__global__ void k_pool_init(float* g) {
  int i = blockIdx.x * blockDim.x + threadIdx.x;
  if (i < N_GRAPHS * D_H) g[i] = -3.402823466e38f;
}

__device__ __forceinline__ void atomicMaxF(float* addr, float v) {
  if (v >= 0.f) atomicMax((int*)addr, __float_as_int(v));
  else          atomicMin((unsigned int*)addr, __float_as_uint(v));
}

__global__ void k_pool_max(const float* __restrict__ h, const int* __restrict__ batch,
                           float* __restrict__ g) {
  const int node = (blockIdx.x * blockDim.x + threadIdx.x) >> 5;
  const int lane = threadIdx.x & 31;
  if (node >= N_NODES) return;
  const int b = batch[node];
  const size_t hb = (size_t)node * D_H;
  float* grow = g + (size_t)b * D_H;
#pragma unroll
  for (int j = 0; j < 8; ++j) {
    const int c = lane + j * 32;
    atomicMaxF(grow + c, h[hb + c]);
  }
}

// ---------------------------------------------------------------------------
// Final 256 -> 4 projection (tiny; one thread per output)
// ---------------------------------------------------------------------------
__global__ void k_cls_out(const float* __restrict__ z, const float* __restrict__ W2,
                          const float* __restrict__ b2, float* __restrict__ out) {
  const int t = threadIdx.x;
  if (t >= N_GRAPHS * N_CLASSES) return;
  const int i = t >> 2, j = t & 3;
  float s = b2[j];
  for (int k = 0; k < D_H; ++k) s += z[i * D_H + k] * W2[k * N_CLASSES + j];
  out[i * N_CLASSES + j] = s;
}

// ---------------------------------------------------------------------------
extern "C" void kernel_launch(void* const* d_in, const int* in_sizes, int n_in,
                              void* d_out, int out_size, void* d_ws, size_t ws_size,
                              hipStream_t stream) {
  (void)in_sizes; (void)n_in; (void)out_size; (void)ws_size;
  const float* x     = (const float*)d_in[0];
  const int*   ei    = (const int*)d_in[1];
  const int*   batch = (const int*)d_in[2];
  const float* W_in  = (const float*)d_in[3];
  const float* b_in  = (const float*)d_in[4];
  const float* convW = (const float*)d_in[5];
  const float* convB = (const float*)d_in[6];
  const float* lnG   = (const float*)d_in[7];
  const float* lnB   = (const float*)d_in[8];
  const float* clsW1 = (const float*)d_in[9];
  const float* clsB1 = (const float*)d_in[10];
  const float* clsW2 = (const float*)d_in[11];
  const float* clsB2 = (const float*)d_in[12];
  const int* src = ei;
  const int* dst = ei + N_EDGES;

  char* ws = (char*)d_ws;
  size_t off = 0;
  auto alloc = [&](size_t bytes) -> char* {
    char* p = ws + off;
    off = (off + bytes + 255) & ~(size_t)255;
    return p;
  };
  float* h    = (float*)alloc((size_t)N_NODES * D_H * 4);
  float* m    = (float*)alloc((size_t)N_NODES * D_H * 4);
  float* acc  = (float*)alloc((size_t)N_NODES * D_H * 4);
  float* deg  = (float*)alloc((size_t)N_NODES * 4);
  float* dinv = (float*)alloc((size_t)N_NODES * 4);
  unsigned short* Wt_in   = (unsigned short*)alloc((size_t)D_H * D_IN * 2);
  unsigned short* Wt_conv = (unsigned short*)alloc((size_t)N_LAYERS * D_H * D_H * 2);
  unsigned short* Wt_cls1 = (unsigned short*)alloc((size_t)D_H * D_H * 2);
  float* gpool = (float*)alloc((size_t)N_GRAPHS * D_H * 4);
  float* z     = (float*)alloc((size_t)N_GRAPHS * D_H * 4);

  // Weights -> bf16, transposed to N x K
  k_transpose_bf16<<<(D_IN * D_H + 255) / 256, 256, 0, stream>>>(W_in, Wt_in, D_IN, D_H);
  for (int i = 0; i < N_LAYERS; ++i)
    k_transpose_bf16<<<(D_H * D_H + 255) / 256, 256, 0, stream>>>(
        convW + (size_t)i * D_H * D_H, Wt_conv + (size_t)i * D_H * D_H, D_H, D_H);
  k_transpose_bf16<<<(D_H * D_H + 255) / 256, 256, 0, stream>>>(clsW1, Wt_cls1, D_H, D_H);

  // Degrees / symmetric norm
  k_deg_init <<<(N_NODES + 255) / 256, 256, 0, stream>>>(deg);
  k_deg_edges<<<(N_EDGES + 255) / 256, 256, 0, stream>>>(dst, deg);
  k_dinv     <<<(N_NODES + 255) / 256, 256, 0, stream>>>(deg, dinv);

  // Input projection: h = x @ W_in + b_in   (M=50000, K=768)
  k_gemm_bf16_n256<<<N_NODES / 16, 128, 0, stream>>>(x, Wt_in, b_in, h, D_IN, 0);

  for (int i = 0; i < N_LAYERS; ++i) {
    // m = h @ convW[i]
    k_gemm_bf16_n256<<<N_NODES / 16, 128, 0, stream>>>(
        h, Wt_conv + (size_t)i * D_H * D_H, nullptr, m, D_H, 0);
    // acc = convB[i] (broadcast)
    k_acc_init<<<(N_NODES * D_H + 255) / 256, 256, 0, stream>>>(
        convB + (size_t)i * D_H, acc);
    // acc += scatter(m[src] * norm -> dst)
    k_aggregate<<<(N_EDGES * 32 + 255) / 256, 256, 0, stream>>>(src, dst, dinv, m, acc);
    // h = LN(relu(acc + m*dinv^2 (self-loop) + h))
    k_combine_ln<<<(N_NODES + 7) / 8, 256, 0, stream>>>(
        acc, m, dinv, h, lnG + (size_t)i * D_H, lnB + (size_t)i * D_H);
  }

  // Global max pool
  k_pool_init<<<(N_GRAPHS * D_H + 255) / 256, 256, 0, stream>>>(gpool);
  k_pool_max <<<(N_NODES * 32 + 255) / 256, 256, 0, stream>>>(h, batch, gpool);

  // z = relu(g @ clsW1 + clsB1)   (M=64, K=256)
  k_gemm_bf16_n256<<<N_GRAPHS / 16, 128, 0, stream>>>(gpool, Wt_cls1, clsB1, z, D_H, 1);
  // out = z @ clsW2 + clsB2
  k_cls_out<<<1, 256, 0, stream>>>(z, clsW2, clsB2, (float*)d_out);
}
